// MyModel_87522843559315
// MI455X (gfx1250) — matrix operand order repaired and checked by reading, not compiled, not run on hardware
//
#include <hip/hip_runtime.h>
#include <stdint.h>

typedef __attribute__((ext_vector_type(16))) __bf16 v16bf;
typedef __attribute__((ext_vector_type(8)))  float  v8f;

typedef unsigned short u16;
typedef unsigned int   u32;

__device__ __forceinline__ u16 f2bf(float f) {
    u32 u = __builtin_bit_cast(u32, f);
    u32 r = u + 0x7FFFu + ((u >> 16) & 1u);   // round-to-nearest-even
    return (u16)(r >> 16);
}
__device__ __forceinline__ float bf2f(u16 h) {
    u32 u = ((u32)h) << 16;
    return __builtin_bit_cast(float, u);
}

struct B256 { uint4 lo, hi; };
__device__ __forceinline__ v16bf pack16(uint4 lo, uint4 hi) {
    B256 t{lo, hi};
    return __builtin_bit_cast(v16bf, t);
}

__device__ __forceinline__ float sigmoidf_(float x) {
    return 1.0f / (1.0f + __expf(-x));
}

// ---------------------------------------------------------------------------
// Prep kernels: put all GEMM operands into bf16 in their final layouts so the
// GEMM needs no LDS staging and no conversions on the hot path.
// ---------------------------------------------------------------------------
__global__ void f32_to_bf16(const float* __restrict__ src, u16* __restrict__ dst, int n) {
    int i = blockIdx.x * blockDim.x + threadIdx.x;
    if (i < n) dst[i] = f2bf(src[i]);
}

// src f32 row-major [K][N] -> dst bf16 [N][K] (K contiguous per column; with this code's k = 16h + s order each
// lane's 16-deep K fragment is one contiguous 32B chunk).
__global__ void transpose_to_bf16(const float* __restrict__ src, u16* __restrict__ dst,
                                  int K, int N) {
    int idx = blockIdx.x * blockDim.x + threadIdx.x;
    if (idx >= K * N) return;
    int n = idx / K, k = idx % K;
    dst[idx] = f2bf(src[(size_t)k * N + n]);
}

__global__ void sum_bf16(const u16* __restrict__ a, const u16* __restrict__ b,
                         u16* __restrict__ dst, int n) {
    int i = blockIdx.x * blockDim.x + threadIdx.x;
    if (i < n) dst[i] = f2bf(bf2f(a[i]) + bf2f(b[i]));
}

__global__ void add_bf16_to_f32(const u16* __restrict__ a, const u16* __restrict__ b,
                                float* __restrict__ out, int n) {
    int i = blockIdx.x * blockDim.x + threadIdx.x;
    if (i < n) out[i] = bf2f(a[i]) + bf2f(b[i]);
}

// ---------------------------------------------------------------------------
// LDS-free streaming GEMM: C = A[M,K](bf16) @ WT[N,K](bf16, pre-transposed)
// + bias[N].  Output written in scan-native layout xz[T][N][B] so the
// recurrent kernel's accumulator seeds are single b128 loads.
// Block 256 thr = 8 independent waves (4 M-tiles x 2 N-groups), block tile
// 64x128, K-panel 32 per v_wmma_f32_16x16x32_bf16. All K-panel loads are
// issued before the WMMA chain so the compiler can use partial loadcnt waits.
// ---------------------------------------------------------------------------
__global__ __launch_bounds__(256)
void gemm_wmma_direct(const u16* __restrict__ A, const u16* __restrict__ WT,
                      const float* __restrict__ bias, u16* __restrict__ C,
                      int M, int N, int K, int T, int Bt) {
    const int tid  = threadIdx.x;
    const int lane = tid & 31;
    const int wave = tid >> 5;
    const int nc   = lane & 15;
    const int hi   = lane >> 4;          // lanes 16-31 hold the upper K half
    const int mt   = wave >> 1;          // 0..3
    const int ng   = wave & 1;           // 0..1

    const int m0 = blockIdx.y * 64 + mt * 16;
    const int n0 = blockIdx.x * 128 + ng * 64;

    v8f acc[4] = {};

    // A fragment: row = m0 + (lane&15); lanes<16: K 0-15; lanes>=16: K 16-31
    const u16* Ap = A + (size_t)(m0 + nc) * K + hi * 16;
    // B fragment: col fixed per lane; lanes<16: K 0-15; lanes>=16: K 16-31
    const u16* Bp0 = WT + (size_t)(n0 +  0 + nc) * K + hi * 16;
    const u16* Bp1 = WT + (size_t)(n0 + 16 + nc) * K + hi * 16;
    const u16* Bp2 = WT + (size_t)(n0 + 32 + nc) * K + hi * 16;
    const u16* Bp3 = WT + (size_t)(n0 + 48 + nc) * K + hi * 16;

    #pragma unroll 2
    for (int k0 = 0; k0 < K; k0 += 32) {
        // issue every load of this K-panel first (in-order returns -> partial waits)
        uint4 alo = *reinterpret_cast<const uint4*>(Ap + k0);
        uint4 ahi = *reinterpret_cast<const uint4*>(Ap + k0 + 8);
        uint4 b0l = *reinterpret_cast<const uint4*>(Bp0 + k0);
        uint4 b0h = *reinterpret_cast<const uint4*>(Bp0 + k0 + 8);
        uint4 b1l = *reinterpret_cast<const uint4*>(Bp1 + k0);
        uint4 b1h = *reinterpret_cast<const uint4*>(Bp1 + k0 + 8);
        uint4 b2l = *reinterpret_cast<const uint4*>(Bp2 + k0);
        uint4 b2h = *reinterpret_cast<const uint4*>(Bp2 + k0 + 8);
        uint4 b3l = *reinterpret_cast<const uint4*>(Bp3 + k0);
        uint4 b3h = *reinterpret_cast<const uint4*>(Bp3 + k0 + 8);
        v16bf a = pack16(alo, ahi);
        acc[0] = __builtin_amdgcn_wmma_f32_16x16x32_bf16(
            false, a, false, pack16(b0l, b0h), (short)0, acc[0], false, false);
        acc[1] = __builtin_amdgcn_wmma_f32_16x16x32_bf16(
            false, a, false, pack16(b1l, b1h), (short)0, acc[1], false, false);
        acc[2] = __builtin_amdgcn_wmma_f32_16x16x32_bf16(
            false, a, false, pack16(b2l, b2h), (short)0, acc[2], false, false);
        acc[3] = __builtin_amdgcn_wmma_f32_16x16x32_bf16(
            false, a, false, pack16(b3l, b3h), (short)0, acc[3], false, false);
    }

    // Epilogue: bias + store to xz[T][N][Bt] (b constant within a 64-row block)
    #pragma unroll
    for (int nt = 0; nt < 4; ++nt) {
        int col = n0 + nt * 16 + nc;
        float bvv = bias[col];
        #pragma unroll
        for (int v = 0; v < 8; ++v) {
            int row = m0 + v + hi * 8;   // C/D layout: lanes>=16 hold M=v+8
            int b = row / T, t = row % T;
            C[((size_t)t * N + col) * Bt + b] = f2bf(acc[nt][v] + bvv);
        }
    }
    (void)M;
}

// ---------------------------------------------------------------------------
// Masked LSTM scan. Batches are independent in the recurrence ->
// grid(4 batch groups of 16 rows, 2 directions). Block: 16 waves; wave w owns
// unit blocks {w, w+16, ...}. h (bf16) in LDS; c in registers in the WMMA C/D
// accumulator layout. z = xz[t] + h @ rT via chained 16x16x32 bf16 WMMA,
// fully unrolled over K. xz layout [T][4U][64] -> seeds are single b128 loads.
// ---------------------------------------------------------------------------
template<int U>
__global__ __launch_bounds__(512)
void lstm_scan(const u16* __restrict__ xz_f, const u16* __restrict__ xz_b,
               const u16* __restrict__ rT_f, const u16* __restrict__ rT_b,
               u16* __restrict__ ho_f, u16* __restrict__ ho_b,
               const int* __restrict__ sizes, int T) {
    constexpr int UBW = U / 256;          // unit blocks per wave (16 waves)
    __shared__ u16 hs[16 * U];            // h, bf16, [row 0..15][u]
    __shared__ int szs[16];

    const int bg  = blockIdx.x;           // 16-row batch group
    const int dir = blockIdx.y;
    const u16* xz = dir ? xz_b : xz_f;
    const u16* rT = dir ? rT_b : rT_f;
    u16*       ho = dir ? ho_b : ho_f;

    const int tid  = threadIdx.x;
    const int lane = tid & 31;
    const int wave = tid >> 5;            // 0..15
    const int nc   = lane & 15;
    const int hi   = lane >> 4;

    for (int i = tid; i < 16 * U; i += 512) hs[i] = 0;
    if (tid < 16) szs[tid] = sizes[bg * 16 + tid];
    __syncthreads();

    float c[UBW][8];
    #pragma unroll
    for (int ub = 0; ub < UBW; ++ub)
        #pragma unroll
        for (int v = 0; v < 8; ++v) c[ub][v] = 0.0f;

    for (int s = 0; s < T; ++s) {
        const int t = dir ? (T - 1 - s) : s;
        const u16* xzt = xz + (size_t)t * (4 * U) * 64;
        float hnew[UBW][8];

        #pragma unroll
        for (int ub = 0; ub < UBW; ++ub) {
            const int u0 = (wave + ub * 16) * 16;
            v8f acc[4];
            // seed accumulators: one b128 per gate (8 batch rows contiguous)
            #pragma unroll
            for (int g = 0; g < 4; ++g) {
                size_t base = (size_t)(g * U + u0 + nc) * 64 + bg * 16 + hi * 8;
                uint4 q = *reinterpret_cast<const uint4*>(&xzt[base]);
                acc[g][0] = __builtin_bit_cast(float, q.x << 16);
                acc[g][1] = __builtin_bit_cast(float, q.x & 0xFFFF0000u);
                acc[g][2] = __builtin_bit_cast(float, q.y << 16);
                acc[g][3] = __builtin_bit_cast(float, q.y & 0xFFFF0000u);
                acc[g][4] = __builtin_bit_cast(float, q.z << 16);
                acc[g][5] = __builtin_bit_cast(float, q.z & 0xFFFF0000u);
                acc[g][6] = __builtin_bit_cast(float, q.w << 16);
                acc[g][7] = __builtin_bit_cast(float, q.w & 0xFFFF0000u);
            }
            // z += h @ r : issue the whole K-panel's loads before the WMMA chain
            const u16* Rp0 = rT + (size_t)(0 * U + u0 + nc) * U + hi * 16;
            const u16* Rp1 = rT + (size_t)(1 * U + u0 + nc) * U + hi * 16;
            const u16* Rp2 = rT + (size_t)(2 * U + u0 + nc) * U + hi * 16;
            const u16* Rp3 = rT + (size_t)(3 * U + u0 + nc) * U + hi * 16;
            #pragma unroll
            for (int k0 = 0; k0 < U; k0 += 32) {
                int abase = nc * U + k0 + hi * 16;
                uint4 alo = *reinterpret_cast<const uint4*>(&hs[abase]);
                uint4 ahi = *reinterpret_cast<const uint4*>(&hs[abase + 8]);
                uint4 b0l = *reinterpret_cast<const uint4*>(Rp0 + k0);
                uint4 b0h = *reinterpret_cast<const uint4*>(Rp0 + k0 + 8);
                uint4 b1l = *reinterpret_cast<const uint4*>(Rp1 + k0);
                uint4 b1h = *reinterpret_cast<const uint4*>(Rp1 + k0 + 8);
                uint4 b2l = *reinterpret_cast<const uint4*>(Rp2 + k0);
                uint4 b2h = *reinterpret_cast<const uint4*>(Rp2 + k0 + 8);
                uint4 b3l = *reinterpret_cast<const uint4*>(Rp3 + k0);
                uint4 b3h = *reinterpret_cast<const uint4*>(Rp3 + k0 + 8);
                v16bf a = pack16(alo, ahi);
                acc[0] = __builtin_amdgcn_wmma_f32_16x16x32_bf16(
                    false, a, false, pack16(b0l, b0h), (short)0, acc[0], false, false);
                acc[1] = __builtin_amdgcn_wmma_f32_16x16x32_bf16(
                    false, a, false, pack16(b1l, b1h), (short)0, acc[1], false, false);
                acc[2] = __builtin_amdgcn_wmma_f32_16x16x32_bf16(
                    false, a, false, pack16(b2l, b2h), (short)0, acc[2], false, false);
                acc[3] = __builtin_amdgcn_wmma_f32_16x16x32_bf16(
                    false, a, false, pack16(b3l, b3h), (short)0, acc[3], false, false);
            }
            // gates + masked state update (Keras: masked steps carry h/c)
            #pragma unroll
            for (int v = 0; v < 8; ++v) {
                int bl = v + hi * 8;
                float iv = sigmoidf_(acc[0][v]);
                float fv = sigmoidf_(acc[1][v]);
                float gv = tanhf(acc[2][v]);
                float ov = sigmoidf_(acc[3][v]);
                float cn = fv * c[ub][v] + iv * gv;
                float hn = ov * tanhf(cn);
                bool  m  = t < szs[bl];
                float hp = bf2f(hs[bl * U + u0 + nc]);
                c[ub][v]    = m ? cn : c[ub][v];
                hnew[ub][v] = m ? hn : hp;
            }
        }
        __syncthreads();   // all waves done reading old h
        #pragma unroll
        for (int ub = 0; ub < UBW; ++ub) {
            const int u0 = (wave + ub * 16) * 16;
            #pragma unroll
            for (int v = 0; v < 8; ++v) {
                int bl = v + hi * 8;
                int b  = bg * 16 + bl;
                u16 hv = f2bf(hnew[ub][v]);
                hs[bl * U + u0 + nc] = hv;
                ho[((size_t)b * T + t) * U + u0 + nc] = hv;
            }
        }
        __syncthreads();   // writes visible before next step's reads
    }
}

// ---------------------------------------------------------------------------
extern "C" void kernel_launch(void* const* d_in, const int* in_sizes, int n_in,
                              void* d_out, int out_size, void* d_ws, size_t ws_size,
                              hipStream_t stream) {
    constexpr int B = 64, T = 512, D = 512, UE = 256, UD = 512;
    const float* seq     = (const float*)d_in[0];
    const int*   sizes   = (const int*)d_in[1];
    const float* enc_kf  = (const float*)d_in[2];
    const float* enc_rf  = (const float*)d_in[3];
    const float* enc_bf_ = (const float*)d_in[4];
    const float* enc_kb  = (const float*)d_in[5];
    const float* enc_rb  = (const float*)d_in[6];
    const float* enc_bb  = (const float*)d_in[7];
    const float* dec_kf  = (const float*)d_in[8];
    const float* dec_rf  = (const float*)d_in[9];
    const float* dec_bf_ = (const float*)d_in[10];
    const float* dec_kb  = (const float*)d_in[11];
    const float* dec_rb  = (const float*)d_in[12];
    const float* dec_bb  = (const float*)d_in[13];

    char* ws = (char*)d_ws;
    size_t off = 0;
    auto alloc = [&](size_t bytes) {
        char* p = ws + off;
        off += (bytes + 255) & ~(size_t)255;
        return p;
    };

    const size_t BT = (size_t)B * T;  // 32768
    // xz region sized for the larger (decoder) layer; reused across layers.
    u16* xzA   = (u16*)alloc(BT * 4 * UD * sizeof(u16));
    u16* xzB   = (u16*)alloc(BT * 4 * UD * sizeof(u16));
    u16* hf    = (u16*)alloc(BT * UE * sizeof(u16));
    u16* hb    = (u16*)alloc(BT * UE * sizeof(u16));
    u16* h2f   = (u16*)alloc(BT * UD * sizeof(u16));
    u16* h2b   = (u16*)alloc(BT * UD * sizeof(u16));
    u16* seqbf = (u16*)alloc(BT * D * sizeof(u16));
    u16* y1    = (u16*)alloc(BT * UE * sizeof(u16));
    u16* rTef  = (u16*)alloc((size_t)4 * UE * UE * sizeof(u16));
    u16* rTeb  = (u16*)alloc((size_t)4 * UE * UE * sizeof(u16));
    u16* rTdf  = (u16*)alloc((size_t)4 * UD * UD * sizeof(u16));
    u16* rTdb  = (u16*)alloc((size_t)4 * UD * UD * sizeof(u16));
    u16* kTef  = (u16*)alloc((size_t)D * 4 * UE * sizeof(u16));
    u16* kTeb  = (u16*)alloc((size_t)D * 4 * UE * sizeof(u16));
    u16* kTdf  = (u16*)alloc((size_t)UE * 4 * UD * sizeof(u16));
    u16* kTdb  = (u16*)alloc((size_t)UE * 4 * UD * sizeof(u16));

    // 0) operand prep: bf16 conversions + weight transposes (all tiny vs 23.3 TB/s)
    {
        int n = (int)(BT * D);
        f32_to_bf16<<<(n + 255) / 256, 256, 0, stream>>>(seq, seqbf, n);
        int ne = 4 * UE * UE, nd = 4 * UD * UD;
        transpose_to_bf16<<<(ne + 255) / 256, 256, 0, stream>>>(enc_rf, rTef, UE, 4 * UE);
        transpose_to_bf16<<<(ne + 255) / 256, 256, 0, stream>>>(enc_rb, rTeb, UE, 4 * UE);
        transpose_to_bf16<<<(nd + 255) / 256, 256, 0, stream>>>(dec_rf, rTdf, UD, 4 * UD);
        transpose_to_bf16<<<(nd + 255) / 256, 256, 0, stream>>>(dec_rb, rTdb, UD, 4 * UD);
        int nke = D * 4 * UE, nkd = UE * 4 * UD;
        transpose_to_bf16<<<(nke + 255) / 256, 256, 0, stream>>>(enc_kf, kTef, D, 4 * UE);
        transpose_to_bf16<<<(nke + 255) / 256, 256, 0, stream>>>(enc_kb, kTeb, D, 4 * UE);
        transpose_to_bf16<<<(nkd + 255) / 256, 256, 0, stream>>>(dec_kf, kTdf, UE, 4 * UD);
        transpose_to_bf16<<<(nkd + 255) / 256, 256, 0, stream>>>(dec_kb, kTdb, UE, 4 * UD);
    }
    // 1) encoder input projections (LDS-free streaming WMMA GEMMs)
    {
        dim3 grid(4 * UE / 128, (unsigned)(BT / 64));
        gemm_wmma_direct<<<grid, 256, 0, stream>>>(seqbf, kTef, enc_bf_, xzA,
                                                   (int)BT, 4 * UE, D, T, B);
        gemm_wmma_direct<<<grid, 256, 0, stream>>>(seqbf, kTeb, enc_bb, xzB,
                                                   (int)BT, 4 * UE, D, T, B);
    }
    // 2) encoder recurrent scans (fwd + bwd concurrently, batch-split)
    lstm_scan<UE><<<dim3(4, 2), 512, 0, stream>>>(xzA, xzB, rTef, rTeb, hf, hb, sizes, T);
    // 3) layer-1 directional sum, then decoder input projections
    {
        int n = (int)(BT * UE);
        sum_bf16<<<(n + 255) / 256, 256, 0, stream>>>(hf, hb, y1, n);
        dim3 grid(4 * UD / 128, (unsigned)(BT / 64));
        gemm_wmma_direct<<<grid, 256, 0, stream>>>(y1, kTdf, dec_bf_, xzA,
                                                   (int)BT, 4 * UD, UE, T, B);
        gemm_wmma_direct<<<grid, 256, 0, stream>>>(y1, kTdb, dec_bb, xzB,
                                                   (int)BT, 4 * UD, UE, T, B);
    }
    // 4) decoder recurrent scans
    lstm_scan<UD><<<dim3(4, 2), 512, 0, stream>>>(xzA, xzB, rTdf, rTdb, h2f, h2b, sizes, T);
    // 5) final directional sum -> f32 output [B,T,512]
    {
        int n = (int)(BT * UD);
        add_bf16_to_f32<<<(n + 255) / 256, 256, 0, stream>>>(h2f, h2b, (float*)d_out, n);
    }
    (void)in_sizes; (void)n_in; (void)out_size; (void)ws_size;
}
